// MLPEncoder_1116691497580
// MI455X (gfx1250) — matrix operand order repaired and checked by reading, not compiled, not run on hardware
//
#include <hip/hip_runtime.h>

// ---------------------------------------------------------------------------
// NRI MLP-Encoder forward for MI455X (gfx1250), bf16 WMMA + f32 accum/stats.
// B=32, N=64, E=4032, NH=256. Output [B,E,2] f32.
// ---------------------------------------------------------------------------

typedef __bf16 bf16;
typedef __attribute__((ext_vector_type(16))) __bf16 v16bf;
typedef __attribute__((ext_vector_type(8)))  float  v8f;
typedef __attribute__((ext_vector_type(4)))  int    v4i;

union BFrag { v16bf v; v4i i[2]; };

#define NHID 256
#define NB   32
#define NN   64
#define NE   4032
#define ROWS_EDGE (NB * NE)   // 129024
#define ROWS_NODE (NB * NN)   // 2048

// ---------------------------------------------------------------------------
// Generic tiled GEMM:  out[M,256] = ELU(A[M,K] @ W[K,256] + bias)
// Wt is W transposed: [256, K] row-major bf16 (B-fragment friendly).
// GATHER: 0 = dense A0[M,K]
//         2 = A row r -> concat( x[b,send(e)] , x[b,recv(e)] )        (K=512)
//         3 = A row r -> concat( x3[b,send], x3[b,recv], xskip[r] )   (K=768)
// Block: 256 threads (8 waves). Each wave: 16 rows x 64 cols (4 wmma accs).
// Block tile: 128 rows x 64 cols.  grid = (256/64, M/128).
// Inner loop: preload all 4 B fragments, then issue 4 back-to-back WMMAs
// (independent accumulators -> no inter-WMMA hazard waits).
// ---------------------------------------------------------------------------
template <int K, int GATHER, bool OUTF32>
__global__ void __launch_bounds__(256)
k_gemm(const bf16* __restrict__ A0, const bf16* __restrict__ A1,
       const bf16* __restrict__ Wt, const float* __restrict__ bias,
       float* __restrict__ outF, bf16* __restrict__ outB)
{
    __shared__ bf16 sw[64 * 256];   // weight chunk: 64 cols x 256 k, 32 KB

    const int tid  = threadIdx.x;
    const int wave = tid >> 5;
    const int lane = tid & 31;
    const int lh   = lane & 15;
    const bool hi  = lane >= 16;

    const int col0 = blockIdx.x * 64;
    const int row0 = blockIdx.y * 128 + wave * 16;
    const int r    = row0 + lh;             // A row for this lane (lanes l, l+16 share)

    // Per-lane base pointers for each 256-wide K segment of the (virtual) A row.
    const bf16 *s0, *s1, *s2;
    if (GATHER == 0) {
        s0 = A0 + (size_t)r * K;
        s1 = s0 + 256;
        s2 = s0 + 512;
    } else {
        const int b  = r / NE;
        const int e  = r - b * NE;
        const int i  = e / 63;              // sender node
        const int jr = e - i * 63;
        const int j  = jr + (jr >= i ? 1 : 0);  // receiver node
        s0 = A0 + (size_t)((b << 6) + i) * NHID;   // sender features
        s1 = A0 + (size_t)((b << 6) + j) * NHID;   // receiver features
        s2 = (GATHER == 3) ? (A1 + (size_t)r * NHID) : s0;  // x_skip per edge
    }

    // ISA lane layouts (bf16, 16x16x32):
    //  A frag: lanes 0-15 hold K {0..7, 16..23}, lanes 16-31 hold K {8..15, 24..31}
    //  B frag: lanes 0-15 hold K {0..7, 16..23}, lanes 16-31 hold K {8..15, 24..31}
    const int aoff = hi ? 8 : 0;
    const int boff = hi ? 8 : 0;

    v8f acc[4];
#pragma unroll
    for (int t = 0; t < 4; ++t)
        acc[t] = v8f{0.f, 0.f, 0.f, 0.f, 0.f, 0.f, 0.f, 0.f};

#pragma unroll
    for (int ci = 0; ci < K / 256; ++ci) {
        const int kc = ci * 256;
        const bf16* abase = (ci == 0) ? s0 : (ci == 1 ? s1 : s2);

        __syncthreads();
        // Cooperative stage of the 64x256 weight chunk into LDS (v4i = 8 bf16).
        {
            v4i* sw4 = (v4i*)sw;
#pragma unroll
            for (int it = 0; it < 8; ++it) {
                const int cidx = tid + it * 256;       // 2048 16B chunks
                const int nl   = cidx >> 5;            // 0..63 (column)
                const int kb   = (cidx & 31) << 3;     // 0..248 (k element)
                sw4[cidx] = *(const v4i*)(Wt + (size_t)(col0 + nl) * K + kc + kb);
            }
        }
        __syncthreads();

        // Prefetch next chunk's A data (global_prefetch_b8 on gfx1250).
        if (ci + 1 < K / 256) {
            const bf16* nb = (GATHER == 0) ? (s0 + kc + 256)
                                           : (ci == 0 ? s1 : s2);
            __builtin_prefetch(nb + aoff, 0, 1);
        }

#pragma unroll
        for (int kk = 0; kk < 256; kk += 32) {
            // A fragment (global, fused gather base)
            BFrag af;
            af.i[0] = *(const v4i*)(abase + kk + aoff);
            af.i[1] = *(const v4i*)(abase + kk + aoff + 16);

            // Preload ALL B fragments for this K-step, then fire the 4 WMMAs
            // back-to-back so the matrix pipe isn't gated per-instruction on
            // s_wait_dscnt.
            BFrag bfr[4];
#pragma unroll
            for (int nt = 0; nt < 4; ++nt) {
                const bf16* bb = sw + (size_t)(nt * 16 + lh) * 256 + kk + boff;
                bfr[nt].i[0] = *(const v4i*)(bb);
                bfr[nt].i[1] = *(const v4i*)(bb + 16);
            }
#pragma unroll
            for (int nt = 0; nt < 4; ++nt) {
                acc[nt] = __builtin_amdgcn_wmma_f32_16x16x32_bf16(
                    false, af.v, false, bfr[nt].v, (short)0, acc[nt], false, false);
            }
        }
    }

    // Epilogue: bias + ELU, store f32 (pre-BN hidden) or bf16 (next GEMM input).
#pragma unroll
    for (int nt = 0; nt < 4; ++nt) {
        const int c  = col0 + nt * 16 + lh;
        const float bv = bias[c];
#pragma unroll
        for (int v = 0; v < 8; ++v) {
            const int rr = row0 + v + (hi ? 8 : 0);
            float val = acc[nt][v] + bv;
            val = val > 0.f ? val : (__expf(val) - 1.f);
            if (OUTF32) outF[(size_t)rr * NHID + c] = val;
            else        outB[(size_t)rr * NHID + c] = (bf16)val;
        }
    }
}

// ---------------------------------------------------------------------------
// Small helper kernels
// ---------------------------------------------------------------------------
__global__ void k_zero(float* p, int n) {
    int i = blockIdx.x * 256 + threadIdx.x;
    if (i < n) p[i] = 0.f;
}

__global__ void k_cvt(const float* __restrict__ s, bf16* __restrict__ d, int n) {
    int i = blockIdx.x * 256 + threadIdx.x;
    if (i < n) d[i] = (bf16)s[i];
}

// W [K, Nc] f32  ->  Wt [Nc, K] bf16
__global__ void k_wt(const float* __restrict__ w, bf16* __restrict__ wt, int K, int Nc) {
    int idx = blockIdx.x * 256 + threadIdx.x;
    if (idx >= K * Nc) return;
    int n = idx / K;
    int k = idx - n * K;
    wt[idx] = (bf16)w[k * Nc + n];
}

// Per-channel sum / sumsq over rows (for BatchNorm batch stats).
__global__ void k_reduce(const float* __restrict__ h, float* __restrict__ st, int M) {
    const int c    = threadIdx.x;
    const int row0 = blockIdx.x * 256;
    const int rend = (row0 + 256 < M) ? (row0 + 256) : M;
    float s = 0.f, s2 = 0.f;
    for (int rrow = row0; rrow < rend; ++rrow) {
        float v = h[(size_t)rrow * NHID + c];
        s += v; s2 += v * v;
    }
    atomicAdd(&st[c], s);
    atomicAdd(&st[NHID + c], s2);
}

// Normalize (training-mode batch stats, biased var) -> bf16 for next GEMM.
__global__ void k_bn_bf16(const float* __restrict__ h, const float* __restrict__ st,
                          const float* __restrict__ g, const float* __restrict__ beta,
                          bf16* __restrict__ out, float invM) {
    const int c    = threadIdx.x;
    const size_t row = blockIdx.x;
    const float mean = st[c] * invM;
    const float var  = st[NHID + c] * invM - mean * mean;
    const float rstd = rsqrtf(var + 1e-5f);
    const float v = h[row * NHID + c];
    out[row * NHID + c] = (bf16)((v - mean) * rstd * g[c] + beta[c]);
}

// edge2node: inc[b,n,c] = (1/N) * sum over edges e with recv(e)==n of xskip[b,e,c]
__global__ void k_edge2node(const bf16* __restrict__ xs, bf16* __restrict__ inc) {
    const int c  = threadIdx.x;
    const int bn = blockIdx.x;          // b*64 + n
    const int b  = bn >> 6;
    const int n  = bn & 63;
    float s = 0.f;
    for (int i = 0; i < NN; ++i) {
        if (i == n) continue;
        const int jr = (n < i) ? n : (n - 1);
        const int e  = i * 63 + jr;
        s += (float)xs[((size_t)b * NE + e) * NHID + c];
    }
    inc[(size_t)bn * NHID + c] = (bf16)(s * (1.0f / 64.0f));
}

// Final BN (mlp4) fused with fc 256->2.  One wave32 per output row.
__global__ void __launch_bounds__(256)
k_bn_fc(const float* __restrict__ h, const float* __restrict__ st,
        const float* __restrict__ g, const float* __restrict__ beta,
        const float* __restrict__ fcw, const float* __restrict__ fcb,
        float* __restrict__ out, float invM) {
    const int tid  = threadIdx.x;
    const int lane = tid & 31;
    const int wave = tid >> 5;
    const size_t row = (size_t)blockIdx.x * 8 + wave;
    float a0 = 0.f, a1 = 0.f;
#pragma unroll
    for (int t = 0; t < 8; ++t) {
        const int c = lane + t * 32;
        const float mean = st[c] * invM;
        const float var  = st[NHID + c] * invM - mean * mean;
        const float rstd = rsqrtf(var + 1e-5f);
        const float xn = (h[row * NHID + c] - mean) * rstd * g[c] + beta[c];
        a0 += xn * fcw[c * 2 + 0];
        a1 += xn * fcw[c * 2 + 1];
    }
    // wave32 tree reduction (warpSize == 32 on gfx1250)
    for (int off = 16; off > 0; off >>= 1) {
        a0 += __shfl_down(a0, off, 32);
        a1 += __shfl_down(a1, off, 32);
    }
    if (lane == 0) {
        out[row * 2 + 0] = a0 + fcb[0];
        out[row * 2 + 1] = a1 + fcb[1];
    }
}

// ---------------------------------------------------------------------------
// Host launcher
// ---------------------------------------------------------------------------
extern "C" void kernel_launch(void* const* d_in, const int* in_sizes, int n_in,
                              void* d_out, int out_size, void* d_ws, size_t ws_size,
                              hipStream_t stream) {
    (void)in_sizes; (void)n_in; (void)out_size; (void)ws_size;

    const float* inputs = (const float*)d_in[0];
    // d_in[1]=rel_rec, d_in[2]=rel_send: one-hot matrices, indices computed analytically.
    const float* m1w1 = (const float*)d_in[3],  *m1b1 = (const float*)d_in[4];
    const float* m1w2 = (const float*)d_in[5],  *m1b2 = (const float*)d_in[6];
    const float* m1g  = (const float*)d_in[7],  *m1be = (const float*)d_in[8];
    const float* m2w1 = (const float*)d_in[9],  *m2b1 = (const float*)d_in[10];
    const float* m2w2 = (const float*)d_in[11], *m2b2 = (const float*)d_in[12];
    const float* m2g  = (const float*)d_in[13], *m2be = (const float*)d_in[14];
    const float* m3w1 = (const float*)d_in[15], *m3b1 = (const float*)d_in[16];
    const float* m3w2 = (const float*)d_in[17], *m3b2 = (const float*)d_in[18];
    const float* m3g  = (const float*)d_in[19], *m3be = (const float*)d_in[20];
    const float* m4w1 = (const float*)d_in[21], *m4b1 = (const float*)d_in[22];
    const float* m4w2 = (const float*)d_in[23], *m4b2 = (const float*)d_in[24];
    const float* m4g  = (const float*)d_in[25], *m4be = (const float*)d_in[26];
    const float* fcw  = (const float*)d_in[27], *fcb  = (const float*)d_in[28];

    // --- workspace carve-out -------------------------------------------------
    char* ws = (char*)d_ws;
    size_t off = 0;
    auto alloc = [&](size_t bytes) -> void* {
        void* p = ws + off;
        off = (off + bytes + 255) & ~(size_t)255;
        return p;
    };
    const size_t NODE_E = (size_t)ROWS_NODE * NHID;   //   524,288 elems
    const size_t EDGE_E = (size_t)ROWS_EDGE * NHID;   // 33,030,144 elems

    float* stats  = (float*)alloc(4 * 2 * NHID * sizeof(float)); // 4 MLPs x (sum,sumsq)
    bf16* w11t = (bf16*)alloc(256 * 256 * 2);
    bf16* w12t = (bf16*)alloc(256 * 256 * 2);
    bf16* w21t = (bf16*)alloc(512 * 256 * 2);
    bf16* w22t = (bf16*)alloc(256 * 256 * 2);
    bf16* w31t = (bf16*)alloc(256 * 256 * 2);
    bf16* w32t = (bf16*)alloc(256 * 256 * 2);
    bf16* w41t = (bf16*)alloc(768 * 256 * 2);
    bf16* w42t = (bf16*)alloc(256 * 256 * 2);
    bf16*  a_node  = (bf16*) alloc(NODE_E * 2);
    bf16*  h1_node = (bf16*) alloc(NODE_E * 2);
    float* h2_node = (float*)alloc(NODE_E * 4);
    bf16*  xnode   = (bf16*) alloc(NODE_E * 2);
    bf16*  x3node  = (bf16*) alloc(NODE_E * 2);
    bf16*  incb    = (bf16*) alloc(NODE_E * 2);
    bf16*  h1_edge = (bf16*) alloc(EDGE_E * 2);
    float* h2_edge = (float*)alloc(EDGE_E * 4);
    bf16*  xskip   = (bf16*) alloc(EDGE_E * 2);

    float* outF = (float*)d_out;

    const float invN = 1.0f / (float)ROWS_NODE;
    const float invE = 1.0f / (float)ROWS_EDGE;
    const dim3 gNode(4, ROWS_NODE / 128);   // (4, 16)
    const dim3 gEdge(4, ROWS_EDGE / 128);   // (4, 1008)

    // --- prep ----------------------------------------------------------------
    k_zero<<<(4 * 2 * NHID + 255) / 256, 256, 0, stream>>>(stats, 4 * 2 * NHID);
    k_cvt <<<(int)((NODE_E + 255) / 256), 256, 0, stream>>>(inputs, a_node, (int)NODE_E);
    k_wt<<<(256 * 256 + 255) / 256, 256, 0, stream>>>(m1w1, w11t, 256, 256);
    k_wt<<<(256 * 256 + 255) / 256, 256, 0, stream>>>(m1w2, w12t, 256, 256);
    k_wt<<<(512 * 256 + 255) / 256, 256, 0, stream>>>(m2w1, w21t, 512, 256);
    k_wt<<<(256 * 256 + 255) / 256, 256, 0, stream>>>(m2w2, w22t, 256, 256);
    k_wt<<<(256 * 256 + 255) / 256, 256, 0, stream>>>(m3w1, w31t, 256, 256);
    k_wt<<<(256 * 256 + 255) / 256, 256, 0, stream>>>(m3w2, w32t, 256, 256);
    k_wt<<<(768 * 256 + 255) / 256, 256, 0, stream>>>(m4w1, w41t, 768, 256);
    k_wt<<<(256 * 256 + 255) / 256, 256, 0, stream>>>(m4w2, w42t, 256, 256);

    // --- mlp1 (nodes) --------------------------------------------------------
    k_gemm<256, 0, false><<<gNode, 256, 0, stream>>>(a_node,  nullptr, w11t, m1b1, nullptr, h1_node);
    k_gemm<256, 0, true ><<<gNode, 256, 0, stream>>>(h1_node, nullptr, w12t, m1b2, h2_node, nullptr);
    k_reduce<<<ROWS_NODE / 256, 256, 0, stream>>>(h2_node, stats + 0 * 512, ROWS_NODE);
    k_bn_bf16<<<ROWS_NODE, 256, 0, stream>>>(h2_node, stats + 0 * 512, m1g, m1be, xnode, invN);

    // --- mlp2 (edges, fused node2edge gather) --------------------------------
    k_gemm<512, 2, false><<<gEdge, 256, 0, stream>>>(xnode,   nullptr, w21t, m2b1, nullptr, h1_edge);
    k_gemm<256, 0, true ><<<gEdge, 256, 0, stream>>>(h1_edge, nullptr, w22t, m2b2, h2_edge, nullptr);
    k_reduce<<<ROWS_EDGE / 256, 256, 0, stream>>>(h2_edge, stats + 1 * 512, ROWS_EDGE);
    k_bn_bf16<<<ROWS_EDGE, 256, 0, stream>>>(h2_edge, stats + 1 * 512, m2g, m2be, xskip, invE);

    // --- edge2node scatter-sum ----------------------------------------------
    k_edge2node<<<ROWS_NODE, 256, 0, stream>>>(xskip, incb);

    // --- mlp3 (nodes) --------------------------------------------------------
    k_gemm<256, 0, false><<<gNode, 256, 0, stream>>>(incb,    nullptr, w31t, m3b1, nullptr, h1_node);
    k_gemm<256, 0, true ><<<gNode, 256, 0, stream>>>(h1_node, nullptr, w32t, m3b2, h2_node, nullptr);
    k_reduce<<<ROWS_NODE / 256, 256, 0, stream>>>(h2_node, stats + 2 * 512, ROWS_NODE);
    k_bn_bf16<<<ROWS_NODE, 256, 0, stream>>>(h2_node, stats + 2 * 512, m3g, m3be, x3node, invN);

    // --- mlp4 (edges, fused [send, recv, skip] gather) -----------------------
    k_gemm<768, 3, false><<<gEdge, 256, 0, stream>>>(x3node,  xskip,   w41t, m4b1, nullptr, h1_edge);
    k_gemm<256, 0, true ><<<gEdge, 256, 0, stream>>>(h1_edge, nullptr, w42t, m4b2, h2_edge, nullptr);
    k_reduce<<<ROWS_EDGE / 256, 256, 0, stream>>>(h2_edge, stats + 3 * 512, ROWS_EDGE);

    // --- final BN + fc (256 -> 2), one wave per edge row ---------------------
    k_bn_fc<<<ROWS_EDGE / 8, 256, 0, stream>>>(h2_edge, stats + 3 * 512, m4g, m4be,
                                               fcw, fcb, outF, invE);
}
